// Attention_47528108097749
// MI455X (gfx1250) — matrix operand order repaired and checked by reading, not compiled, not run on hardware
//
#include <hip/hip_runtime.h>

// ---------------------------------------------------------------------------
// CDNA5 / gfx1250 attention (LLaMA GQA prefill), bf16 WMMA + f32 accumulate.
// All matrix tiles staged by the Tensor Data Mover (double-buffered, DMA
// overlapped with WMMA compute). B=2, T=2048, D=4096, H=32, KV=8, HD=128.
// ---------------------------------------------------------------------------

typedef __attribute__((ext_vector_type(16))) __bf16 v16bf;
typedef __attribute__((ext_vector_type(8)))  __bf16 v8bf;
typedef __attribute__((ext_vector_type(8)))  float  v8f;
typedef __attribute__((ext_vector_type(4)))  unsigned int u32x4;
typedef __attribute__((ext_vector_type(8)))  unsigned int u32x8;

#define WMMA_BF16(a, b, c) \
  __builtin_amdgcn_wmma_f32_16x16x32_bf16(false, (a), false, (b), (short)0, (c), false, false)

// Issue a 2-D TDM tile load (group0 built here, group1 supplied by caller).
// D# group0: [1:0]=count=1, [63:32]=lds_addr, [120:64]=global_addr, [127:126]=type 2.
__device__ __forceinline__ void tdm_load_2d(unsigned int lds_off, const void* gaddr,
                                            u32x8 g1) {
  unsigned long long ga = (unsigned long long)gaddr;
  u32x4 g0;
  g0.x = 1u;                                                   // count = 1
  g0.y = lds_off;                                              // LDS byte address
  g0.z = (unsigned int)ga;                                     // global_addr[31:0]
  g0.w = (unsigned int)((ga >> 32) & 0x1FFFFFFu) | (2u << 30); // addr[56:32], type=2
  asm volatile("tensor_load_to_lds %0, %1" :: "s"(g0), "s"(g1) : "memory");
}

// TDM group1 builder: data_size=2B, pad enabled (padi/pada are the ISA codes),
// 2-D tile dim0(x) x dim1(y), row stride in elements.
__device__ __forceinline__ u32x8 tdm_g1(unsigned int dim0, unsigned int dim1,
                                        unsigned int stride, unsigned int padi,
                                        unsigned int pada) {
  u32x8 g;
  g[0] = 0x00010000u | (1u << 20) | (padi << 22) | (pada << 25);
  g[1] = dim0 << 16;         // tensor_dim0[15:0]
  g[2] = dim1 << 16;         // tensor_dim0 hi=0, tensor_dim1[15:0]
  g[3] = dim0 << 16;         // tensor_dim1 hi=0, tile_dim0
  g[4] = dim1;               // tile_dim1 (tile_dim2=0)
  g[5] = stride;             // tensor_dim0_stride[31:0]
  g[6] = 0u;
  g[7] = 0u;
  return g;
}

// ------------------------- elementwise prep kernels ------------------------

__global__ __launch_bounds__(256) void convert_x_kernel(const float* __restrict__ x,
                                                        __bf16* __restrict__ xb, int n) {
  int i = blockIdx.x * 256 + threadIdx.x;
  if (i < n) xb[i] = (__bf16)x[i];
}

// W f32 [N,K] -> W^T bf16 [K,N]  (LDS-tiled 32x32 transpose, coalesced both sides)
__global__ __launch_bounds__(256) void transpose_w_kernel(const float* __restrict__ W,
                                                          __bf16* __restrict__ WT,
                                                          int N, int K) {
  __shared__ __bf16 tile[32][33];
  const int n0 = blockIdx.x * 32, k0 = blockIdx.y * 32;
  const int tx = threadIdx.x, ty = threadIdx.y;      // (32, 8)
#pragma unroll
  for (int j = 0; j < 32; j += 8)
    tile[ty + j][tx] = (__bf16)W[(size_t)(n0 + ty + j) * K + k0 + tx];
  __syncthreads();
#pragma unroll
  for (int j = 0; j < 32; j += 8)
    WT[(size_t)(k0 + ty + j) * N + n0 + tx] = tile[tx][ty + j];
}

// q f32 [B,T,32,128] -> RoPE*scale -> qb bf16 [B,32,T,128]
__global__ __launch_bounds__(256) void rope_q_kernel(const float* __restrict__ qf,
                                                     const float* __restrict__ cosT,
                                                     const float* __restrict__ sinT,
                                                     __bf16* __restrict__ qb) {
  int idx = blockIdx.x * 256 + threadIdx.x;      // B*T*32*64 = 8388608
  int i = idx & 63;
  int h = (idx >> 6) & 31;
  int t = (idx >> 11) & 2047;
  int b = idx >> 22;
  float c = cosT[t * 64 + i], s = sinT[t * 64 + i];
  size_t src = (((size_t)(b * 2048 + t) * 32 + h) * 128) + 2 * i;
  float re = qf[src], im = qf[src + 1];
  const float scale = 0.08838834764831845f;      // 1/sqrt(128), folded into Q
  size_t dst = (((size_t)(b * 32 + h) * 2048 + t) * 128) + 2 * i;
  qb[dst]     = (__bf16)((re * c - im * s) * scale);
  qb[dst + 1] = (__bf16)((re * s + im * c) * scale);
}

// k f32 [B,T,8,128] -> RoPE -> K^T bf16 [B,8,128,2048]
__global__ __launch_bounds__(256) void rope_k_kernel(const float* __restrict__ kf,
                                                     const float* __restrict__ cosT,
                                                     const float* __restrict__ sinT,
                                                     __bf16* __restrict__ kbT) {
  int idx = blockIdx.x * 256 + threadIdx.x;      // B*T*8*64 = 2097152
  int i = idx & 63;
  int kh = (idx >> 6) & 7;
  int t = (idx >> 9) & 2047;
  int b = idx >> 20;
  float c = cosT[t * 64 + i], s = sinT[t * 64 + i];
  size_t src = (((size_t)(b * 2048 + t) * 8 + kh) * 128) + 2 * i;
  float re = kf[src], im = kf[src + 1];
  size_t dst = ((size_t)(b * 8 + kh) * 128 + 2 * i) * 2048 + t;
  kbT[dst]        = (__bf16)(re * c - im * s);
  kbT[dst + 2048] = (__bf16)(re * s + im * c);
}

// v f32 [B,T,8,128] -> vb bf16 [B,8,T,128]
__global__ __launch_bounds__(256) void pack_v_kernel(const float* __restrict__ vf,
                                                     __bf16* __restrict__ vb) {
  int idx = blockIdx.x * 256 + threadIdx.x;      // B*T*8*128 = 4194304
  int d = idx & 127;
  int kh = (idx >> 7) & 7;
  int t = (idx >> 10) & 2047;
  int b = idx >> 21;
  size_t src = (((size_t)(b * 2048 + t) * 8 + kh) * 128) + d;
  size_t dst = (((size_t)(b * 8 + kh) * 2048 + t) * 128) + d;
  vb[dst] = (__bf16)vf[src];
}

// ----------------------------- WMMA GEMM -----------------------------------
// C[M,N] (f32) = A_bf16[M,K] * WT_bf16[K,N].  Block: 256 thr = 8 waves,
// tile 128x128, K-step 32. Both tiles TDM-staged, double-buffered so the DMA
// for step i+1 overlaps the WMMAs of step i.
__global__ __launch_bounds__(256) void gemm_bf16_kernel(const __bf16* __restrict__ A,
                                                        const __bf16* __restrict__ WT,
                                                        float* __restrict__ C,
                                                        int N, int K) {
  __shared__ __bf16 lA[2][128][40];   // [m][k], 32 data + 8 pad (TDM pads)
  __shared__ __bf16 lB[2][32][136];   // [k][n], 128 data + 8 pad (TDM pads)
  const int tid = threadIdx.x;
  const int lane = tid & 31;
  const int w = tid >> 5;
  const int wm = w & 3;               // 4 row groups of 32
  const int wn = w >> 2;              // 2 col groups of 64
  const int m0 = blockIdx.y * 128;
  const int n0 = blockIdx.x * 128;
  const int rA = lane & 15;
  const int ko = (lane >> 4) * 8;

  const u32x8 gA = tdm_g1(32, 128, (unsigned)K, 3, 3);   // 16dw rows + 4dw pad
  const u32x8 gB = tdm_g1(128, 32, (unsigned)N, 5, 3);   // 64dw rows + 4dw pad
  const unsigned int ldsA = (unsigned int)(unsigned long long)&lA[0][0][0];
  const unsigned int ldsB = (unsigned int)(unsigned long long)&lB[0][0][0];
  const __bf16* Ab = A + (size_t)m0 * K;

  if (tid < 32) {                     // prologue: fill buffer 0
    tdm_load_2d(ldsA, Ab, gA);
    tdm_load_2d(ldsB, WT + n0, gB);
  }

  v8f acc[2][4] = {};

  for (int k0 = 0; k0 < K; k0 += 32) {
    const int buf = (k0 >> 5) & 1;
    __builtin_amdgcn_s_wait_tensorcnt(0);
    __syncthreads();                  // buf ready; prev compute done everywhere
    if (tid < 32 && k0 + 32 < K) {    // launch next chunk into other buffer
      tdm_load_2d(ldsA + (buf ^ 1) * (unsigned)sizeof(lA[0]),
                  Ab + k0 + 32, gA);
      tdm_load_2d(ldsB + (buf ^ 1) * (unsigned)sizeof(lB[0]),
                  WT + (size_t)(k0 + 32) * N + n0, gB);
    }

    v16bf a[2], b[4];
#pragma unroll
    for (int mi = 0; mi < 2; ++mi) {
      int r = wm * 32 + mi * 16 + rA;
      union { v16bf v; v8bf h2[2]; } u;
      u.h2[0] = *(const v8bf*)&lA[buf][r][ko];
      u.h2[1] = *(const v8bf*)&lA[buf][r][16 + ko];
      a[mi] = u.v;
    }
#pragma unroll
    for (int ni = 0; ni < 4; ++ni) {
      int nb = wn * 64 + ni * 16;
      union { v16bf v; v8bf h2[2]; } u;
#pragma unroll
      for (int s = 0; s < 16; ++s)
        u.v[s] = lB[buf][(s < 8 ? ko + s : 8 + ko + s)][nb + rA];
      b[ni] = u.v;
    }
#pragma unroll
    for (int mi = 0; mi < 2; ++mi)
#pragma unroll
      for (int ni = 0; ni < 4; ++ni)
        acc[mi][ni] = WMMA_BF16(a[mi], b[ni], acc[mi][ni]);
    __syncthreads();                  // compute(buf) done before its next DMA
  }

  // D layout: VGPR r, lane l -> m = r + 8*(l>>4), n = l&15
  const int mh = (lane >> 4) * 8;
  const int nl = lane & 15;
#pragma unroll
  for (int mi = 0; mi < 2; ++mi)
#pragma unroll
    for (int ni = 0; ni < 4; ++ni)
#pragma unroll
      for (int r = 0; r < 8; ++r)
        C[(size_t)(m0 + wm * 32 + mi * 16 + mh + r) * N +
          (n0 + wn * 64 + ni * 16 + nl)] = acc[mi][ni][r];
}

// --------------------------- flash attention -------------------------------
// grid (T/64, H, B), block 128 (4 waves); wave w owns q rows [tq0, tq0+16).
// qb [B,32,T,128], kbT [B,8,128,2048], vb [B,8,T,128];
// out ctx bf16 [B,T,32,128]. K^T and V tiles TDM-staged, double-buffered:
// chunk i+1 DMA overlaps chunk i scores/softmax/PV.
__global__ __launch_bounds__(128) void attn_kernel(const __bf16* __restrict__ qb,
                                                   const __bf16* __restrict__ kbT,
                                                   const __bf16* __restrict__ vb,
                                                   __bf16* __restrict__ ctxb) {
  __shared__ __bf16 sKt[2][128][40];  // [d][s], 32 data + 8 pad (TDM pads)
  __shared__ __bf16 sV[2][32][136];   // [s][d], 128 data + 8 pad (TDM pads)
  __shared__ __bf16 sP[4][16][32];    // wave-private P tiles [t][s]
  const int tid = threadIdx.x;
  const int lane = tid & 31;
  const int w = tid >> 5;
  const int h = blockIdx.y;
  const int b = blockIdx.z;
  const int kh = h >> 2;              // GQA: 4 q heads per kv head
  const int tq0 = blockIdx.x * 64 + w * 16;
  const __bf16* qptr = qb + (size_t)(b * 32 + h) * 2048 * 128;
  const __bf16* ktp  = kbT + (size_t)(b * 8 + kh) * 128 * 2048;
  const __bf16* vptr = vb + (size_t)(b * 8 + kh) * 2048 * 128;
  const int rA = lane & 15;
  const int ko = (lane >> 4) * 8;

  const u32x8 gK = tdm_g1(32, 128, 2048u, 3, 3);   // K^T tile: 32(s) x 128(d)
  const u32x8 gV = tdm_g1(128, 32, 128u, 5, 3);    // V tile: 128(d) x 32(s)
  const unsigned int ldsK = (unsigned int)(unsigned long long)&sKt[0][0][0];
  const unsigned int ldsV = (unsigned int)(unsigned long long)&sV[0][0][0];

  if (w == 0) {                       // prologue: fill buffer 0
    tdm_load_2d(ldsK, ktp, gK);
    tdm_load_2d(ldsV, vptr, gV);
  }

  // Q fragments (scaled by 1/sqrt(HD) at rope time): 4 K-chunks of 32
  v16bf aQ[4];
#pragma unroll
  for (int dc = 0; dc < 4; ++dc) {
    const __bf16* p = qptr + (size_t)(tq0 + rA) * 128 + dc * 32;
    union { v16bf v; v8bf h2[2]; } u;
    u.h2[0] = *(const v8bf*)(p + ko);
    u.h2[1] = *(const v8bf*)(p + 16 + ko);
    aQ[dc] = u.v;
  }

  v8f acc[8] = {};
  float m_i[8], l_i[8];
#pragma unroll
  for (int r = 0; r < 8; ++r) { m_i[r] = -1e30f; l_i[r] = 0.0f; }

  for (int s0 = 0; s0 < 2048; s0 += 32) {
    const int buf = (s0 >> 5) & 1;
    __builtin_amdgcn_s_wait_tensorcnt(0);
    __syncthreads();                  // buf ready; prev compute done everywhere
    if (w == 0 && s0 + 32 < 2048) {   // launch next chunk into other buffer
      tdm_load_2d(ldsK + (buf ^ 1) * (unsigned)sizeof(sKt[0]), ktp + s0 + 32, gK);
      tdm_load_2d(ldsV + (buf ^ 1) * (unsigned)sizeof(sV[0]),
                  vptr + (size_t)(s0 + 32) * 128, gV);
    }

    // scores: S[16x32] = Q[16x128] * K^T  (B-frag: lane=N(s), elements=K(d))
    v8f sc0 = {}, sc1 = {};
#pragma unroll
    for (int dc = 0; dc < 4; ++dc) {
      union { v16bf v; v8bf h2[2]; } u0, u1;
#pragma unroll
      for (int s = 0; s < 16; ++s) {
        int kk = dc * 32 + (s < 8 ? ko + s : 8 + ko + s);
        u0.v[s] = sKt[buf][kk][rA];
        u1.v[s] = sKt[buf][kk][16 + rA];
      }
      sc0 = WMMA_BF16(aQ[dc], u0.v, sc0);
      sc1 = WMMA_BF16(aQ[dc], u1.v, sc1);
    }

    // online softmax per row (row m = r + 8*(lane>>4); reduce over 16 lanes)
    float alpha[8];
#pragma unroll
    for (int r = 0; r < 8; ++r) {
      float mx = fmaxf(sc0[r], sc1[r]);
      mx = fmaxf(mx, __shfl_xor(mx, 1, 32));
      mx = fmaxf(mx, __shfl_xor(mx, 2, 32));
      mx = fmaxf(mx, __shfl_xor(mx, 4, 32));
      mx = fmaxf(mx, __shfl_xor(mx, 8, 32));
      float nm = fmaxf(m_i[r], mx);
      float p0 = __expf(sc0[r] - nm);
      float p1 = __expf(sc1[r] - nm);
      float al = __expf(m_i[r] - nm);
      m_i[r] = nm;
      alpha[r] = al;
      float rs = p0 + p1;
      rs += __shfl_xor(rs, 1, 32);
      rs += __shfl_xor(rs, 2, 32);
      rs += __shfl_xor(rs, 4, 32);
      rs += __shfl_xor(rs, 8, 32);
      l_i[r] = l_i[r] * al + rs;
      int tl = r + 8 * (lane >> 4);
      sP[w][tl][lane & 15] = (__bf16)p0;        // D-layout -> [t][s] in LDS
      sP[w][tl][16 + (lane & 15)] = (__bf16)p1;
    }
#pragma unroll
    for (int df = 0; df < 8; ++df)
#pragma unroll
      for (int r = 0; r < 8; ++r) acc[df][r] *= alpha[r];

    // reload P in A-frag layout (wave-private LDS round trip)
    union { v16bf v; v8bf h2[2]; } up;
    up.h2[0] = *(const v8bf*)&sP[w][rA][ko];
    up.h2[1] = *(const v8bf*)&sP[w][rA][16 + ko];
    v16bf aP = up.v;

    // ctx += P[16x32] * V[32x128]  (B-frag: lane=N(d), elements=K(s))
#pragma unroll
    for (int df = 0; df < 8; ++df) {
      union { v16bf v; v8bf h2[2]; } uv;
#pragma unroll
      for (int s = 0; s < 16; ++s)
        uv.v[s] = sV[buf][(s < 8 ? ko + s : 8 + ko + s)][df * 16 + rA];
      acc[df] = WMMA_BF16(aP, uv.v, acc[df]);
    }
    __syncthreads();                  // compute(buf) done before its next DMA
  }

  // write ctx / l_i : out[b][t][h][d]
#pragma unroll
  for (int r = 0; r < 8; ++r) {
    float inv = 1.0f / l_i[r];
    int t = tq0 + r + 8 * (lane >> 4);
#pragma unroll
    for (int df = 0; df < 8; ++df) {
      int d = df * 16 + (lane & 15);
      ctxb[(((size_t)(b * 2048 + t) * 32 + h) * 128) + d] =
          (__bf16)(acc[df][r] * inv);
    }
  }
}

// ------------------------------- launcher ----------------------------------

extern "C" void kernel_launch(void* const* d_in, const int* in_sizes, int n_in,
                              void* d_out, int out_size, void* d_ws, size_t ws_size,
                              hipStream_t stream) {
  const float* x    = (const float*)d_in[0];
  const float* wq   = (const float*)d_in[1];
  const float* wk   = (const float*)d_in[2];
  const float* wv   = (const float*)d_in[3];
  const float* wo   = (const float*)d_in[4];
  const float* cosT = (const float*)d_in[7];
  const float* sinT = (const float*)d_in[8];
  float* out = (float*)d_out;
  (void)in_sizes; (void)n_in; (void)out_size; (void)ws_size;

  char* ws = (char*)d_ws;
  size_t off = 0;
  auto carve = [&](size_t bytes) -> char* {
    char* p = ws + off;
    off += (bytes + 255) & ~(size_t)255;
    return p;
  };
  const size_t M = 4096;                        // B*T
  __bf16* xb   = (__bf16*)carve(M * 4096 * 2);  // 32 MB
  float*  qf   = (float*) carve(M * 4096 * 4);  // 64 MB
  float*  kf   = (float*) carve(M * 1024 * 4);  // 16 MB
  float*  vf   = (float*) carve(M * 1024 * 4);  // 16 MB (contiguous after kf)
  __bf16* qbf  = (__bf16*)carve(M * 4096 * 2);  // 32 MB
  __bf16* kbT  = (__bf16*)carve(M * 1024 * 2);  //  8 MB (K transposed)
  __bf16* vbf  = (__bf16*)carve(M * 1024 * 2);  //  8 MB
  __bf16* wqT  = (__bf16*)carve(4096 * 4096 * 2); // 32 MB
  __bf16* wkT  = (__bf16*)carve(4096 * 1024 * 2); //  8 MB
  __bf16* wvT  = (__bf16*)carve(4096 * 1024 * 2); //  8 MB
  __bf16* ctxb = (__bf16*)qf;                   // alias: qf dead after rope_q
  __bf16* woT  = (__bf16*)kf;                   // alias: kf+vf dead after rope/pack

  dim3 tb(32, 8);
  convert_x_kernel<<<16777216 / 256, 256, 0, stream>>>(x, xb, 16777216);
  transpose_w_kernel<<<dim3(128, 128), tb, 0, stream>>>(wq, wqT, 4096, 4096);
  transpose_w_kernel<<<dim3(32, 128), tb, 0, stream>>>(wk, wkT, 1024, 4096);
  transpose_w_kernel<<<dim3(32, 128), tb, 0, stream>>>(wv, wvT, 1024, 4096);

  gemm_bf16_kernel<<<dim3(32, 32), 256, 0, stream>>>(xb, wqT, qf, 4096, 4096);
  gemm_bf16_kernel<<<dim3(8, 32), 256, 0, stream>>>(xb, wkT, kf, 1024, 4096);
  gemm_bf16_kernel<<<dim3(8, 32), 256, 0, stream>>>(xb, wvT, vf, 1024, 4096);

  rope_q_kernel<<<8388608 / 256, 256, 0, stream>>>(qf, cosT, sinT, qbf);
  rope_k_kernel<<<2097152 / 256, 256, 0, stream>>>(kf, cosT, sinT, kbT);
  pack_v_kernel<<<4194304 / 256, 256, 0, stream>>>(vf, vbf);

  transpose_w_kernel<<<dim3(128, 128), tb, 0, stream>>>(wo, woT, 4096, 4096);

  attn_kernel<<<dim3(32, 32, 2), 128, 0, stream>>>(qbf, kbT, vbf, ctxb);

  gemm_bf16_kernel<<<dim3(32, 32), 256, 0, stream>>>(ctxb, woT, out, 4096, 4096);
}
